// WaveNet_70059506532597
// MI455X (gfx1250) — compile-verified, matrix operand order checked by reading, not run on hardware
//
#include <hip/hip_runtime.h>

typedef __bf16 bf16;
typedef __attribute__((ext_vector_type(16))) __bf16 v16bf;
typedef __attribute__((ext_vector_type(8)))  float  v8f;
typedef __attribute__((ext_vector_type(4)))  unsigned int u32x4;
typedef __attribute__((ext_vector_type(8)))  int i32x8;
typedef __attribute__((ext_vector_type(4)))  int i32x4;

#define TDIM  32768
#define NB    8
#define CRESC 64
#define CINC  128
#define COUTC 80
#define NLAY  10
#define TT    256            // time tile per block (2 x 16 per wave)
#define TILES (TDIM / TT)    // 128 tiles per batch row

#if __has_builtin(__builtin_amdgcn_tensor_load_to_lds)
#define HAVE_TDM 1
#else
#define HAVE_TDM 0
#endif

// K index inside a 16-bit 16x32 WMMA fragment for (lane, element e) per CDNA5 ISA:
// lanes 0-15 hold K = {0..7}(v0-3),{16..23}(v4-7); lanes 16-31 hold {8..15},{24..31}
__device__ __forceinline__ int kfrag(int lane, int e) {
  return ((lane >> 4) << 3) + e + ((e >> 3) << 3);
}

#if HAVE_TDM
// 1D contiguous global->LDS DMA via Tensor Data Mover (6-arg clang-23 form).
// D# group0: count=1 | lds_addr | global_addr | type=2.  group1: data_size=8B,
// tensor_dim0 = tile_dim0 = nbytes/8, tensor_dim1 = tile_dim1 = 1.
__device__ __forceinline__ void tdm_load_1d(void* lds, const void* g, unsigned nbytes) {
  unsigned lds_off = (unsigned)(unsigned long long)(uintptr_t)lds; // low 32b = LDS offset
  unsigned long long ga = (unsigned long long)(uintptr_t)g;
  unsigned n8 = nbytes >> 3;
  u32x4 g0;
  g0[0] = 1u;                                               // count=1, user D#
  g0[1] = lds_off;                                          // lds_addr (bytes)
  g0[2] = (unsigned)(ga & 0xFFFFFFFFu);                     // global_addr[31:0]
  g0[3] = (unsigned)((ga >> 32) & 0x1FFFFFFu) | (2u << 30); // addr[56:32] | type=2
  i32x8 g1;
  g1[0] = (int)(3u << 16);        // data_size = 3 (8 bytes)
  g1[1] = (int)(n8 << 16);        // tensor_dim0[15:0]  (bits 79:48)
  g1[2] = (int)(1u << 16);        // tensor_dim0 hi = 0, tensor_dim1 = 1
  g1[3] = (int)(n8 << 16);        // tile_dim0 (bits 127:112)
  g1[4] = 1;                      // tile_dim1 = 1, tile_dim2 = 0
  g1[5] = (int)n8;                // tensor_dim0_stride lo
  g1[6] = 0;
  g1[7] = 0;
  i32x4 gz4 = {0, 0, 0, 0};
  i32x8 gz8 = {0, 0, 0, 0, 0, 0, 0, 0};
  __builtin_amdgcn_tensor_load_to_lds(g0, g1, gz4, gz4, gz8, 0);
}
#endif

__device__ __forceinline__ void coop_copy16(bf16* dst, const bf16* src,
                                            int n_bf16, int tid, int nt) {
  const uint4* s = (const uint4*)src;
  uint4* d = (uint4*)dst;
  int n = n_bf16 >> 3;
  for (int i = tid; i < n; i += nt) d[i] = s[i];
}

// -------------------------------------------------------------------------
// Weight pre-swizzle: f32 -> bf16 A-fragments in workspace.
// frag ids: [0,16)    front   : mt*4+kt                    (M=64, K=128)
//           [16,336)  conv    : 16 + L*32 + (gate?16:0) + mt*4 + tap*2 + kt
//           [336,496) res/skip: 336 + L*16 + (skip?8:0) + mt*2 + kt
//           [496,504) out1, [504,514) out2 (M=80 -> 5 m-tiles)
// Fragment: 32 lanes x 16 bf16, lane-major (32B per lane).
// -------------------------------------------------------------------------
__global__ void swz_all(const float* __restrict__ front_w,
                        const float* __restrict__ filter_w,
                        const float* __restrict__ gate_w,
                        const float* __restrict__ res_w,
                        const float* __restrict__ skip_w,
                        const float* __restrict__ out1_w,
                        const float* __restrict__ out2_w,
                        bf16* __restrict__ dst) {
  int f   = blockIdx.x;
  int tid = threadIdx.x;           // 512 threads: lane = tid>>4, e = tid&15
  int l = tid >> 4, e = tid & 15;
  int m16 = l & 15;
  int kk  = kfrag(l, e);
  float v;
  if (f < 16) {
    int mt = f >> 2, kt = f & 3;
    v = front_w[(mt*16 + m16)*CINC + kt*32 + kk];
  } else if (f < 336) {
    int lf = f - 16; int L = lf >> 5; int r = lf & 31;
    const float* W = (r < 16 ? filter_w : gate_w) + L*CRESC*CRESC*2;
    int r16 = r & 15; int mt = r16 >> 2, tap = (r16 >> 1) & 1, kt = r16 & 1;
    v = W[((mt*16 + m16)*CRESC + kt*32 + kk)*2 + tap];
  } else if (f < 496) {
    int lf = f - 336; int L = lf >> 4; int r = lf & 15;
    const float* W = (r < 8 ? res_w : skip_w) + L*CRESC*CRESC;
    int r8 = r & 7; int mt = r8 >> 1, kt = r8 & 1;
    v = W[(mt*16 + m16)*CRESC + kt*32 + kk];
  } else if (f < 504) {
    int r = f - 496; int mt = r >> 1, kt = r & 1;
    v = out1_w[(mt*16 + m16)*CRESC + kt*32 + kk];
  } else {
    int r = f - 504; int mt = r >> 1, kt = r & 1;
    v = out2_w[(mt*16 + m16)*CRESC + kt*32 + kk];
  }
  dst[f*512 + tid] = (bf16)v;
}

// -------------------------------------------------------------------------
// Front 1x1 conv: h = W(64x128) @ x + b, x f32 -> h bf16.
// Each wave handles two 16-wide time tiles.
// -------------------------------------------------------------------------
__global__ __launch_bounds__(256) void front_kernel(
    const float* __restrict__ x, bf16* __restrict__ h,
    const bf16* __restrict__ frags, const float* __restrict__ front_b) {
  __shared__ __align__(32) bf16 lsA[16*512];          // 16 KB of A-fragments
  int tid = threadIdx.x, lane = tid & 31, wave = tid >> 5;
  int b = blockIdx.x / TILES, t0 = (blockIdx.x % TILES) * TT;
  int tcol = lane & 15, hi = lane >> 4;

#if HAVE_TDM
  if (wave == 0) {
    tdm_load_1d(lsA, frags, 16*1024);
    __builtin_amdgcn_s_wait_tensorcnt(0);
  }
#else
  coop_copy16(lsA, frags, 16*512, tid, 256);
#endif
  __syncthreads();

  int tb = t0 + wave*32 + tcol;
  v16bf Bt[2][4];                                     // [half][kt]
#pragma unroll
  for (int hh = 0; hh < 2; ++hh) {
    int ts = tb + hh*16;
#pragma unroll
    for (int kt = 0; kt < 4; ++kt) {
      v16bf bv;
#pragma unroll
      for (int e = 0; e < 16; ++e) {
        int k = kt*32 + kfrag(lane, e);
        bv[e] = (bf16)x[(b*CINC + k)*TDIM + ts];
      }
      Bt[hh][kt] = bv;
    }
  }
#pragma unroll
  for (int j = 0; j < 4; ++j) {
    v8f acc[2] = {{}, {}};
#pragma unroll
    for (int kt = 0; kt < 4; ++kt) {
      v16bf a = *(const v16bf*)(lsA + (j*4 + kt)*512 + lane*16);
#pragma unroll
      for (int hh = 0; hh < 2; ++hh)
        acc[hh] = __builtin_amdgcn_wmma_f32_16x16x32_bf16(false, a, false, Bt[hh][kt],
                                                          (short)0, acc[hh], false, false);
    }
#pragma unroll
    for (int hh = 0; hh < 2; ++hh)
#pragma unroll
      for (int r = 0; r < 8; ++r) {
        int c = j*16 + r + hi*8;                      // D layout: M = r + 8*(lane>>4)
        h[(b*CRESC + c)*TDIM + tb + hh*16] = (bf16)(acc[hh][r] + front_b[c]);
      }
  }
}

// -------------------------------------------------------------------------
// Fused WaveNet layer: dilated conv (filter+gate) -> tanh*sigmoid -> res/skip.
// A-fragments staged in LDS once per block (TDM), shared by all 8 waves.
// -------------------------------------------------------------------------
__global__ __launch_bounds__(256) void layer_kernel(
    const bf16* __restrict__ hin, bf16* __restrict__ hout,
    float* __restrict__ skip, const bf16* __restrict__ frags,
    const float* __restrict__ filter_b, const float* __restrict__ gate_b,
    const float* __restrict__ res_b, const float* __restrict__ skip_b,
    int layer, int dil, float beta) {
  __shared__ __align__(32) bf16 lsA[48*512];          // 48 KB: conv(32) + res/skip(16)
  __shared__ __align__(32) bf16 lsZ[8][2][2][32][16]; // 32 KB per-wave z transpose
  int tid = threadIdx.x, lane = tid & 31, wave = tid >> 5;
  int b = blockIdx.x / TILES, t0 = (blockIdx.x % TILES) * TT;
  int tcol = lane & 15, hi = lane >> 4;

  const bf16* convA = frags + (16 + layer*32)*512;
  const bf16* rsA   = frags + (336 + layer*16)*512;
  const float* fb = filter_b + layer*CRESC;
  const float* gb = gate_b   + layer*CRESC;
  const float* rb = res_b    + layer*CRESC;
  const float* sb = skip_b   + layer*CRESC;

#if HAVE_TDM
  if (wave == 0) {
    tdm_load_1d(lsA, convA, 32*1024);
    tdm_load_1d(lsA + 32*512, rsA, 16*1024);
    __builtin_amdgcn_s_wait_tensorcnt(0);
  }
#else
  coop_copy16(lsA, convA, 32*512, tid, 256);
  coop_copy16(lsA + 32*512, rsA, 16*512, tid, 256);
#endif
  __syncthreads();

  int tb = t0 + wave*32 + tcol;

  // Gather B fragments, two halves x two taps: tap0 -> x[t-d], tap1 -> x[t]
  v16bf Bt[2][2][2];                                  // [half][tap][kt]
#pragma unroll
  for (int hh = 0; hh < 2; ++hh) {
#pragma unroll
    for (int tap = 0; tap < 2; ++tap) {
      int ts = tb + hh*16 - (tap == 0 ? dil : 0);
#pragma unroll
      for (int kt = 0; kt < 2; ++kt) {
        v16bf bv;
#pragma unroll
        for (int e = 0; e < 16; ++e) {
          int k = kt*32 + kfrag(lane, e);
          bv[e] = (ts >= 0) ? hin[(b*CRESC + k)*TDIM + ts] : (bf16)0.0f;
        }
        Bt[hh][tap][kt] = bv;
      }
    }
  }

  // filter/gate GEMMs + gating; z written to LDS in B-fragment layout
#pragma unroll
  for (int j = 0; j < 4; ++j) {
    v8f accF[2] = {{}, {}}, accG[2] = {{}, {}};
#pragma unroll
    for (int tap = 0; tap < 2; ++tap) {
#pragma unroll
      for (int kt = 0; kt < 2; ++kt) {
        v16bf aF = *(const v16bf*)(lsA + (j*4 + tap*2 + kt)*512 + lane*16);
        v16bf aG = *(const v16bf*)(lsA + (16 + j*4 + tap*2 + kt)*512 + lane*16);
#pragma unroll
        for (int hh = 0; hh < 2; ++hh) {
          accF[hh] = __builtin_amdgcn_wmma_f32_16x16x32_bf16(false, aF, false, Bt[hh][tap][kt],
                                                             (short)0, accF[hh], false, false);
          accG[hh] = __builtin_amdgcn_wmma_f32_16x16x32_bf16(false, aG, false, Bt[hh][tap][kt],
                                                             (short)0, accG[hh], false, false);
        }
      }
    }
#pragma unroll
    for (int hh = 0; hh < 2; ++hh)
#pragma unroll
      for (int r = 0; r < 8; ++r) {
        int c  = j*16 + r + hi*8;
        float yf = accF[hh][r] + fb[c];
        float yg = accG[hh][r] + gb[c];
        float z  = tanhf(yf) / (1.0f + __expf(-yg));
        int k32 = c & 31;
        lsZ[wave][hh][c >> 5][tcol + (((k32 >> 3) & 1) << 4)]
            [(k32 & 7) + ((k32 >> 4) << 3)] = (bf16)z;
      }
  }

  // same-wave DS ops are in-order on CDNA5; just block compiler reordering
  asm volatile("" ::: "memory");

  v16bf Z[2][2];
#pragma unroll
  for (int hh = 0; hh < 2; ++hh) {
    Z[hh][0] = *(const v16bf*)&lsZ[wave][hh][0][lane][0];
    Z[hh][1] = *(const v16bf*)&lsZ[wave][hh][1][lane][0];
  }

  // res + skip 1x1 GEMMs, residual add, skip accumulate
#pragma unroll
  for (int j = 0; j < 4; ++j) {
    v8f aR[2] = {{}, {}}, aS[2] = {{}, {}};
    v16bf w0 = *(const v16bf*)(lsA + (32 + j*2 + 0)*512 + lane*16);
    v16bf w1 = *(const v16bf*)(lsA + (32 + j*2 + 1)*512 + lane*16);
    v16bf s0 = *(const v16bf*)(lsA + (32 + 8 + j*2 + 0)*512 + lane*16);
    v16bf s1 = *(const v16bf*)(lsA + (32 + 8 + j*2 + 1)*512 + lane*16);
#pragma unroll
    for (int hh = 0; hh < 2; ++hh) {
      aR[hh] = __builtin_amdgcn_wmma_f32_16x16x32_bf16(false, w0, false, Z[hh][0], (short)0, aR[hh], false, false);
      aR[hh] = __builtin_amdgcn_wmma_f32_16x16x32_bf16(false, w1, false, Z[hh][1], (short)0, aR[hh], false, false);
      aS[hh] = __builtin_amdgcn_wmma_f32_16x16x32_bf16(false, s0, false, Z[hh][0], (short)0, aS[hh], false, false);
      aS[hh] = __builtin_amdgcn_wmma_f32_16x16x32_bf16(false, s1, false, Z[hh][1], (short)0, aS[hh], false, false);
    }
#pragma unroll
    for (int hh = 0; hh < 2; ++hh)
#pragma unroll
      for (int r = 0; r < 8; ++r) {
        int c = j*16 + r + hi*8;
        int idx = (b*CRESC + c)*TDIM + tb + hh*16;
        hout[idx] = (bf16)((float)hin[idx] + aR[hh][r] + rb[c]);
        skip[idx] = beta * skip[idx] + aS[hh][r] + sb[c];
      }
  }
}

// -------------------------------------------------------------------------
// Output head: relu(skip) -> out1(64x64)+relu -> out2(80x64) -> d_out f32
// -------------------------------------------------------------------------
__global__ __launch_bounds__(256) void out_kernel(
    const float* __restrict__ skip, float* __restrict__ out,
    const bf16* __restrict__ frags,
    const float* __restrict__ out1_b, const float* __restrict__ out2_b) {
  __shared__ __align__(32) bf16 lsA[18*512];          // out1(8) + out2(10) frags
  __shared__ __align__(32) bf16 lsZ[8][2][2][32][16];
  int tid = threadIdx.x, lane = tid & 31, wave = tid >> 5;
  int b = blockIdx.x / TILES, t0 = (blockIdx.x % TILES) * TT;
  int tcol = lane & 15, hi = lane >> 4;

#if HAVE_TDM
  if (wave == 0) {
    tdm_load_1d(lsA, frags + 496*512, 18*1024);
    __builtin_amdgcn_s_wait_tensorcnt(0);
  }
#else
  coop_copy16(lsA, frags + 496*512, 18*512, tid, 256);
#endif
  __syncthreads();

  int tb = t0 + wave*32 + tcol;
  v16bf Bt[2][2];
#pragma unroll
  for (int hh = 0; hh < 2; ++hh)
#pragma unroll
    for (int kt = 0; kt < 2; ++kt) {
      v16bf bv;
#pragma unroll
      for (int e = 0; e < 16; ++e) {
        int k = kt*32 + kfrag(lane, e);
        bv[e] = (bf16)fmaxf(skip[(b*CRESC + k)*TDIM + tb + hh*16], 0.0f);
      }
      Bt[hh][kt] = bv;
    }
#pragma unroll
  for (int j = 0; j < 4; ++j) {
    v8f acc[2] = {{}, {}};
    v16bf a0 = *(const v16bf*)(lsA + (j*2 + 0)*512 + lane*16);
    v16bf a1 = *(const v16bf*)(lsA + (j*2 + 1)*512 + lane*16);
#pragma unroll
    for (int hh = 0; hh < 2; ++hh) {
      acc[hh] = __builtin_amdgcn_wmma_f32_16x16x32_bf16(false, a0, false, Bt[hh][0], (short)0, acc[hh], false, false);
      acc[hh] = __builtin_amdgcn_wmma_f32_16x16x32_bf16(false, a1, false, Bt[hh][1], (short)0, acc[hh], false, false);
    }
#pragma unroll
    for (int hh = 0; hh < 2; ++hh)
#pragma unroll
      for (int r = 0; r < 8; ++r) {
        int c = j*16 + r + hi*8;
        float q = fmaxf(acc[hh][r] + out1_b[c], 0.0f);
        int k32 = c & 31;
        lsZ[wave][hh][c >> 5][tcol + (((k32 >> 3) & 1) << 4)]
            [(k32 & 7) + ((k32 >> 4) << 3)] = (bf16)q;
      }
  }
  asm volatile("" ::: "memory");
  v16bf Z[2][2];
#pragma unroll
  for (int hh = 0; hh < 2; ++hh) {
    Z[hh][0] = *(const v16bf*)&lsZ[wave][hh][0][lane][0];
    Z[hh][1] = *(const v16bf*)&lsZ[wave][hh][1][lane][0];
  }
#pragma unroll
  for (int j = 0; j < 5; ++j) {                       // M = 80 -> 5 m-tiles
    v8f acc[2] = {{}, {}};
    v16bf a0 = *(const v16bf*)(lsA + (8 + j*2 + 0)*512 + lane*16);
    v16bf a1 = *(const v16bf*)(lsA + (8 + j*2 + 1)*512 + lane*16);
#pragma unroll
    for (int hh = 0; hh < 2; ++hh) {
      acc[hh] = __builtin_amdgcn_wmma_f32_16x16x32_bf16(false, a0, false, Z[hh][0], (short)0, acc[hh], false, false);
      acc[hh] = __builtin_amdgcn_wmma_f32_16x16x32_bf16(false, a1, false, Z[hh][1], (short)0, acc[hh], false, false);
    }
#pragma unroll
    for (int hh = 0; hh < 2; ++hh)
#pragma unroll
      for (int r = 0; r < 8; ++r) {
        int c = j*16 + r + hi*8;
        out[(b*COUTC + c)*TDIM + tb + hh*16] = acc[hh][r] + out2_b[c];
      }
  }
}

// -------------------------------------------------------------------------
extern "C" void kernel_launch(void* const* d_in, const int* in_sizes, int n_in,
                              void* d_out, int out_size, void* d_ws, size_t ws_size,
                              hipStream_t stream) {
  (void)in_sizes; (void)n_in; (void)out_size; (void)ws_size;
  const float* x        = (const float*)d_in[0];
  const float* front_w  = (const float*)d_in[1];
  const float* front_b  = (const float*)d_in[2];
  const float* filter_w = (const float*)d_in[3];
  const float* filter_b = (const float*)d_in[4];
  const float* gate_w   = (const float*)d_in[5];
  const float* gate_b   = (const float*)d_in[6];
  const float* res_w    = (const float*)d_in[7];
  const float* res_b    = (const float*)d_in[8];
  const float* skip_w   = (const float*)d_in[9];
  const float* skip_b   = (const float*)d_in[10];
  const float* out1_w   = (const float*)d_in[11];
  const float* out1_b   = (const float*)d_in[12];
  const float* out2_w   = (const float*)d_in[13];
  const float* out2_b   = (const float*)d_in[14];

  char* ws = (char*)d_ws;
  size_t hbytes = (size_t)NB * CRESC * TDIM * 2;   // 32 MB bf16 activation buffer
  size_t sbytes = (size_t)NB * CRESC * TDIM * 4;   // 64 MB f32 skip accumulator
  bf16*  hA    = (bf16*)ws;
  bf16*  hB    = (bf16*)(ws + hbytes);
  float* skipb = (float*)(ws + 2*hbytes);
  bf16*  frags = (bf16*)(ws + 2*hbytes + sbytes);  // 514 frags * 1KB

  swz_all<<<514, 512, 0, stream>>>(front_w, filter_w, gate_w, res_w, skip_w,
                                   out1_w, out2_w, frags);

  dim3 grid(NB * TILES), blk(256);
  front_kernel<<<grid, blk, 0, stream>>>(x, hA, frags, front_b);
  for (int i = 0; i < NLAY; ++i) {
    bf16* hin  = (i & 1) ? hB : hA;
    bf16* hout = (i & 1) ? hA : hB;
    layer_kernel<<<grid, blk, 0, stream>>>(hin, hout, skipb, frags,
        filter_b, gate_b, res_b, skip_b, i, 1 << i, (i == 0) ? 0.0f : 1.0f);
  }
  out_kernel<<<grid, blk, 0, stream>>>(skipb, (float*)d_out, frags, out1_b, out2_b);
}
